// Attention_62732292325599
// MI455X (gfx1250) — matrix operand order repaired and checked by reading, not compiled, not run on hardware
//
#include <hip/hip_runtime.h>

#define D_MODEL 1024
#define SEQ     2048
#define BATCH   4
#define NHEADS  16
#define DK      64

#define GLOBAL_AS __attribute__((address_space(1)))
#define LDS_AS    __attribute__((address_space(3)))

typedef __attribute__((ext_vector_type(16))) __bf16 v16bf;
typedef __attribute__((ext_vector_type(8)))  float  v8f;
typedef __attribute__((ext_vector_type(4)))  int    v4i;

union FragU { v16bf v; uint4 u[2]; };

__device__ __forceinline__ unsigned short f2bf(float f) {
    union { float f; unsigned u; } x; x.f = f;
    unsigned r = (x.u + 0x7FFFu + ((x.u >> 16) & 1u)) >> 16;
    return (unsigned short)r;
}

// A fragment (16x32 bf16, M x K): lane holds row (lane&15);
// chunks k in [b0,b0+8) and [b0+16,b0+24), b0 = (lane>=16)?8:0
__device__ __forceinline__ v16bf load_a_frag(const unsigned short* rowptr, int kbase, int lane) {
    const int b0 = (lane & 16) ? 8 : 0;
    FragU f;
    f.u[0] = *reinterpret_cast<const uint4*>(rowptr + kbase + b0);
    f.u[1] = *reinterpret_cast<const uint4*>(rowptr + kbase + b0 + 16);
    return f.v;
}

// B fragment (32x16 bf16, K x N): lane holds column's source row;
// chunks k in [c0,c0+8) and [c0+16,c0+24), c0 = (lane>=16)?8:0
__device__ __forceinline__ v16bf load_b_frag(const unsigned short* rowptr, int kbase, int lane) {
    const int c0 = (lane & 16) ? 8 : 0;
    FragU f;
    f.u[0] = *reinterpret_cast<const uint4*>(rowptr + kbase + c0);
    f.u[1] = *reinterpret_cast<const uint4*>(rowptr + kbase + c0 + 16);
    return f.v;
}

__device__ __forceinline__ v8f wmma_bf16(v16bf a, v16bf b, v8f c) {
    return __builtin_amdgcn_wmma_f32_16x16x32_bf16(false, a, false, b, (short)0, c, false, false);
}

// 16B global -> LDS copy: async (ASYNCcnt / gfx1250 path) when available,
// else synchronous load + ds_store_b128.
__device__ __forceinline__ void cp_g2l_b128(const unsigned short* g, unsigned short* l) {
#if __has_builtin(__builtin_amdgcn_global_load_async_to_lds_b128)
    // integer round-trip to build correctly address-spaced pointers:
    // global generic addresses are bit-identical to AS(1); generic LDS
    // addresses carry the LDS offset in addr[31:0] (ISA aperture rules).
    GLOBAL_AS v4i* gp = (GLOBAL_AS v4i*)(unsigned long long)g;
    LDS_AS    v4i* lp = (LDS_AS v4i*)(unsigned int)(unsigned long long)l;
    __builtin_amdgcn_global_load_async_to_lds_b128(gp, lp, 0, 0);
#else
    *reinterpret_cast<uint4*>(l) = *reinterpret_cast<const uint4*>(g);
#endif
}

__device__ __forceinline__ void async_wait_all() {
#if __has_builtin(__builtin_amdgcn_global_load_async_to_lds_b128)
#if __has_builtin(__builtin_amdgcn_s_wait_asynccnt)
    __builtin_amdgcn_s_wait_asynccnt(0);
#else
    asm volatile("s_wait_asynccnt 0" ::: "memory");
#endif
#endif
}

// ---------------- fp32 -> bf16 convert ----------------
__global__ void f32_to_bf16_kernel(const float* __restrict__ src, unsigned short* __restrict__ dst, int n) {
    int i = blockIdx.x * blockDim.x + threadIdx.x;
    if (i < n) dst[i] = f2bf(src[i]);
}

// ---------------- fused QKV projection + RoPE + V-transpose ----------------
// Each wave computes one 16(M) x 64(N) block for Q, K and V simultaneously.
__global__ void __launch_bounds__(128)
qkv_rope_kernel(const unsigned short* __restrict__ xb,
                const unsigned short* __restrict__ wqb,
                const unsigned short* __restrict__ wkb,
                const unsigned short* __restrict__ wvb,
                const int* __restrict__ pos,
                unsigned short* __restrict__ Qb,
                unsigned short* __restrict__ Kb,
                unsigned short* __restrict__ Vt) {
    const int lane = threadIdx.x & 31;
    const int wid  = threadIdx.x >> 5;
    const int gw   = blockIdx.x * 4 + wid;   // 0..8191
    const int m0   = (gw >> 4) * 16;         // 512 M tiles
    const int n0   = (gw & 15) * 64;         // 16 N blocks

    v8f zero = {0.f,0.f,0.f,0.f,0.f,0.f,0.f,0.f};
    v8f accq[4], acck[4], accv[4];
#pragma unroll
    for (int j = 0; j < 4; ++j) { accq[j] = zero; acck[j] = zero; accv[j] = zero; }

    const unsigned short* arow = xb + (size_t)(m0 + (lane & 15)) * D_MODEL;

    for (int k = 0; k < D_MODEL; k += 32) {
        v16bf a = load_a_frag(arow, k, lane);
#pragma unroll
        for (int j = 0; j < 4; ++j) {
            const int n = n0 + j * 16 + (lane & 15);
            v16bf bq = load_b_frag(wqb + (size_t)n * D_MODEL, k, lane);
            accq[j] = wmma_bf16(a, bq, accq[j]);
            v16bf bk = load_b_frag(wkb + (size_t)n * D_MODEL, k, lane);
            acck[j] = wmma_bf16(a, bk, acck[j]);
            v16bf bv = load_b_frag(wvb + (size_t)n * D_MODEL, k, lane);
            accv[j] = wmma_bf16(a, bv, accv[j]);
        }
    }

    const int half8 = (lane & 16) ? 8 : 0;
#pragma unroll
    for (int j = 0; j < 4; ++j) {
        const int n    = n0 + j * 16 + (lane & 15);
        const int didx = n & (DK - 1);
        const int jj   = didx >> 1;
        // theta^{-2j/dk} = exp(-j * ln(10000)/32)
        const float inv_freq = __expf(-(float)jj * 0.28782313662425574f);
#pragma unroll
        for (int r = 0; r < 8; ++r) {
            const int m = m0 + r + half8;
            const int s = m & (SEQ - 1);
            const float p = (float)pos[s];
            float sn, cs;
            __sincosf(p * inv_freq, &sn, &cs);
            // Q with RoPE (pair partner lives in lane^1: same row, adjacent n)
            {
                float v  = accq[j][r];
                float pr = __shfl_xor(v, 1, 32);
                float o  = (didx & 1) ? (pr * sn + v * cs) : (v * cs - pr * sn);
                Qb[(size_t)m * D_MODEL + n] = f2bf(o);
            }
            // K with RoPE
            {
                float v  = acck[j][r];
                float pr = __shfl_xor(v, 1, 32);
                float o  = (didx & 1) ? (pr * sn + v * cs) : (v * cs - pr * sn);
                Kb[(size_t)m * D_MODEL + n] = f2bf(o);
            }
            // V stored transposed per head: Vt[(b*16+h)*64 + d][s]
            {
                const int bidx = m >> 11;
                const int hh   = n >> 6;
                Vt[((size_t)(bidx * NHEADS + hh) * DK + didx) * SEQ + s] = f2bf(accv[j][r]);
            }
        }
    }
}

// ---------------- causal flash attention (block-cooperative K/V staging) ----------------
// Block = 4 waves = 64 q rows of one (b,h). K tile (32x64) and V tile (64x32)
// are staged once per k-block into LDS via async global->LDS loads and shared
// by all 4 waves. Online softmax per wave.
__global__ void __launch_bounds__(128)
flash_attn_kernel(const unsigned short* __restrict__ Qb,
                  const unsigned short* __restrict__ Kb,
                  const unsigned short* __restrict__ Vt,
                  unsigned short* __restrict__ attnb) {
    __shared__ __align__(16) unsigned short ldsK[32 * 64];   // [k-local][d]
    __shared__ __align__(16) unsigned short ldsV[64 * 32];   // [d][k-local]
    __shared__ __align__(16) unsigned short plds[4][16 * 40];

    const int tid  = threadIdx.x;
    const int lane = tid & 31;
    const int wid  = tid >> 5;
    unsigned short* myp = &plds[wid][0];

    const int blk = blockIdx.x;          // 0..2047
    const int qtb = blk & 31;            // 32 q-bands (64 rows) per (b,h)
    const int bh  = blk >> 5;            // b*16 + h
    const int h   = bh & (NHEADS - 1);
    const int b   = bh >> 4;
    const int q0b = qtb * 64;
    const int q0  = q0b + wid * 16;

    const int half8 = (lane & 16) ? 8 : 0;
    const int ncol  = lane & 15;

    const unsigned short* qptr = Qb + ((size_t)(b * SEQ + q0 + ncol)) * D_MODEL + h * DK;
    v16bf aq0 = load_a_frag(qptr, 0, lane);
    v16bf aq1 = load_a_frag(qptr, 32, lane);

    v8f zero = {0.f,0.f,0.f,0.f,0.f,0.f,0.f,0.f};
    v8f o[4];
#pragma unroll
    for (int j = 0; j < 4; ++j) o[j] = zero;
    float mrow[8], lrow[8];
#pragma unroll
    for (int r = 0; r < 8; ++r) { mrow[r] = -3.0e38f; lrow[r] = 0.f; }

    // uniform trip count for the whole block: covers rows up to q0b+63
    const int nk = (q0b + 64) >> 5;      // q0b/32 + 2

    for (int kb = 0; kb < nk; ++kb) {
        const int c0 = kb * 32;

        // ---- cooperative stage: K tile 32 rows x 128B, V tile 64 rows x 64B ----
        {
            int c = tid;                               // K: 256 x 16B chunks
#pragma unroll
            for (int it = 0; it < 2; ++it, c += 128) {
                const int krow = c >> 3, koff = (c & 7) * 8;
                cp_g2l_b128(Kb + ((size_t)(b * SEQ + c0 + krow)) * D_MODEL + h * DK + koff,
                            ldsK + krow * 64 + koff);
            }
            c = tid;                                   // V: 256 x 16B chunks
#pragma unroll
            for (int it = 0; it < 2; ++it, c += 128) {
                const int d = c >> 2, voff = (c & 3) * 8;
                cp_g2l_b128(Vt + ((size_t)bh * DK + d) * SEQ + c0 + voff,
                            ldsV + d * 32 + voff);
            }
        }
        async_wait_all();
        __syncthreads();

        if (c0 < q0 + 16) {                // wave-uniform causal guard (EXEC stays all-1)
            // S = Q K^T for two 16-wide column tiles (K from LDS)
            v8f s0 = zero, s1 = zero;
            {
                const unsigned short* krow0 = ldsK + (size_t)ncol * 64;
                const unsigned short* krow1 = ldsK + (size_t)(16 + ncol) * 64;
                s0 = wmma_bf16(aq0, load_b_frag(krow0, 0, lane),  s0);
                s0 = wmma_bf16(aq1, load_b_frag(krow0, 32, lane), s0);
                s1 = wmma_bf16(aq0, load_b_frag(krow1, 0, lane),  s1);
                s1 = wmma_bf16(aq1, load_b_frag(krow1, 32, lane), s1);
            }
            const bool needmask = (c0 + 31 > q0);
            float e0[8], e1[8];
#pragma unroll
            for (int r = 0; r < 8; ++r) {
                float v0 = s0[r] * 0.125f;       // 1/sqrt(64)
                float v1 = s1[r] * 0.125f;
                if (needmask) {
                    const int m = q0 + r + half8;
                    if (c0 + ncol > m)      v0 = -3.0e38f;
                    if (c0 + 16 + ncol > m) v1 = -3.0e38f;
                }
                float mx = fmaxf(v0, v1);
                mx = fmaxf(mx, __shfl_xor(mx, 1, 32));
                mx = fmaxf(mx, __shfl_xor(mx, 2, 32));
                mx = fmaxf(mx, __shfl_xor(mx, 4, 32));
                mx = fmaxf(mx, __shfl_xor(mx, 8, 32));
                const float mn    = fmaxf(mrow[r], mx);
                const float alpha = __expf(mrow[r] - mn);
                mrow[r] = mn;
                const float p0 = __expf(v0 - mn);
                const float p1 = __expf(v1 - mn);
                float rs = p0 + p1;
                rs += __shfl_xor(rs, 1, 32);
                rs += __shfl_xor(rs, 2, 32);
                rs += __shfl_xor(rs, 4, 32);
                rs += __shfl_xor(rs, 8, 32);
                lrow[r] = lrow[r] * alpha + rs;
                e0[r] = p0; e1[r] = p1;
#pragma unroll
                for (int j = 0; j < 4; ++j) o[j][r] *= alpha;
            }
            // P (16x32) through per-wave LDS slice: C-layout -> A-layout transpose
#pragma unroll
            for (int r = 0; r < 8; ++r) {
                const int m = r + half8;
                myp[m * 40 + ncol]      = f2bf(e0[r]);
                myp[m * 40 + 16 + ncol] = f2bf(e1[r]);
            }
            asm volatile("s_wait_dscnt 0" ::: "memory");   // intra-wave LDS RAW fence
            v16bf pa = load_a_frag(myp + (size_t)ncol * 40, 0, lane);
            asm volatile("" ::: "memory");
            // O += P @ V  (V tile rows are per-head d, contiguous over local k)
#pragma unroll
            for (int j = 0; j < 4; ++j) {
                o[j] = wmma_bf16(pa, load_b_frag(ldsV + (size_t)(j * 16 + ncol) * 32, 0, lane), o[j]);
            }
        }
        __syncthreads();
    }

    // normalize + store merged-head output (bf16)
#pragma unroll
    for (int r = 0; r < 8; ++r) {
        const float inv = 1.0f / lrow[r];
        const int m = q0 + r + half8;
#pragma unroll
        for (int j = 0; j < 4; ++j) {
            const int n = h * DK + j * 16 + ncol;
            attnb[((size_t)(b * SEQ + m)) * D_MODEL + n] = f2bf(o[j][r] * inv);
        }
    }
}

// ---------------- output projection (bf16 in, fp32 out) ----------------
__global__ void __launch_bounds__(128)
out_proj_kernel(const unsigned short* __restrict__ attnb,
                const unsigned short* __restrict__ wob,
                float* __restrict__ out) {
    const int lane = threadIdx.x & 31;
    const int wid  = threadIdx.x >> 5;
    const int gw   = blockIdx.x * 4 + wid;
    const int m0   = (gw >> 4) * 16;
    const int n0   = (gw & 15) * 64;

    v8f zero = {0.f,0.f,0.f,0.f,0.f,0.f,0.f,0.f};
    v8f acc[4];
#pragma unroll
    for (int j = 0; j < 4; ++j) acc[j] = zero;

    const unsigned short* arow = attnb + (size_t)(m0 + (lane & 15)) * D_MODEL;
    for (int k = 0; k < D_MODEL; k += 32) {
        v16bf a = load_a_frag(arow, k, lane);
#pragma unroll
        for (int j = 0; j < 4; ++j) {
            const int n = n0 + j * 16 + (lane & 15);
            acc[j] = wmma_bf16(a, load_b_frag(wob + (size_t)n * D_MODEL, k, lane), acc[j]);
        }
    }
    const int half8 = (lane & 16) ? 8 : 0;
#pragma unroll
    for (int j = 0; j < 4; ++j) {
        const int n = n0 + j * 16 + (lane & 15);
#pragma unroll
        for (int r = 0; r < 8; ++r) {
            const int m = m0 + r + half8;
            out[(size_t)m * D_MODEL + n] = acc[j][r];
        }
    }
}

extern "C" void kernel_launch(void* const* d_in, const int* in_sizes, int n_in,
                              void* d_out, int out_size, void* d_ws, size_t ws_size,
                              hipStream_t stream) {
    (void)in_sizes; (void)n_in; (void)out_size; (void)ws_size;
    const float* x   = (const float*)d_in[0];
    const float* wq  = (const float*)d_in[1];
    const float* wk  = (const float*)d_in[2];
    const float* wv  = (const float*)d_in[3];
    const float* wo  = (const float*)d_in[4];
    const int*   pos = (const int*)d_in[5];

    const size_t NX = (size_t)BATCH * SEQ * D_MODEL;      // 8388608
    const size_t NW = (size_t)D_MODEL * D_MODEL;          // 1048576

    unsigned short* p   = (unsigned short*)d_ws;
    unsigned short* xb  = p;  p += NX;
    unsigned short* wqb = p;  p += NW;
    unsigned short* wkb = p;  p += NW;
    unsigned short* wvb = p;  p += NW;
    unsigned short* wob = p;  p += NW;
    unsigned short* Qb  = p;  p += NX;
    unsigned short* Kb  = p;  p += NX;
    unsigned short* Vt  = p;  p += NX;
    unsigned short* attnb = p;

    f32_to_bf16_kernel<<<(int)((NX + 255) / 256), 256, 0, stream>>>(x,  xb,  (int)NX);
    f32_to_bf16_kernel<<<(int)((NW + 255) / 256), 256, 0, stream>>>(wq, wqb, (int)NW);
    f32_to_bf16_kernel<<<(int)((NW + 255) / 256), 256, 0, stream>>>(wk, wkb, (int)NW);
    f32_to_bf16_kernel<<<(int)((NW + 255) / 256), 256, 0, stream>>>(wv, wvb, (int)NW);
    f32_to_bf16_kernel<<<(int)((NW + 255) / 256), 256, 0, stream>>>(wo, wob, (int)NW);

    qkv_rope_kernel<<<2048, 128, 0, stream>>>(xb, wqb, wkb, wvb, pos, Qb, Kb, Vt);
    flash_attn_kernel<<<2048, 128, 0, stream>>>(Qb, Kb, Vt, attnb);
    out_proj_kernel<<<2048, 128, 0, stream>>>(attnb, wob, (float*)d_out);
}
